// GraphAttentionLayer_34772055228979
// MI455X (gfx1250) — compile-verified, matrix operand order checked by reading, not run on hardware
//
#include <hip/hip_runtime.h>
#include <math.h>

typedef __attribute__((ext_vector_type(16))) _Float16 v16h;
typedef __attribute__((ext_vector_type(8)))  _Float16 v8h;
typedef __attribute__((ext_vector_type(8)))  float    v8f;

#define F          128     // in features == out features (H*C)
#define HEADS      4
#define CH         32
#define NEG_SLOPE  0.2f
#define GAT_EPS    1e-16f
#define PW_ELEMS   (4 * 8 * 32 * 16)   // packed W: 16384 f16 = 32 KB

__device__ __forceinline__ float lrelu(float v) { return v > 0.0f ? v : NEG_SLOPE * v; }
__device__ __forceinline__ unsigned enc_ord(float f) {
    unsigned u = __float_as_uint(f);
    return (u & 0x80000000u) ? ~u : (u | 0x80000000u);
}
__device__ __forceinline__ float dec_ord(unsigned u) {
    return __uint_as_float((u & 0x80000000u) ? (u ^ 0x80000000u) : ~u);
}

// ---------------- phase 0: precision conversion / weight packing ----------------

__global__ void k_cvt_x(const float* __restrict__ x, _Float16* __restrict__ xh, int n) {
    int i = blockIdx.x * blockDim.x + threadIdx.x;
    if (i < n) xh[i] = (_Float16)x[i];
}

// packedW[((kt*8+nt)*32 + lane)*16 + elem] = f16(W[kt*32 + kmap(lane,elem)][nt*16 + (lane&15)])
// kmap matches the 16-bit 16x32 A/B fragment layout (ISA 7.12.2).
__global__ void k_pack_w(const float* __restrict__ W, _Float16* __restrict__ pw) {
    int t = blockIdx.x * blockDim.x + threadIdx.x;     // 0 .. 16383
    if (t >= PW_ELEMS) return;
    int elem = t & 15;
    int lane = (t >> 4) & 31;
    int nt   = (t >> 9) & 7;
    int kt   = t >> 12;
    int b0 = (lane < 16) ? 0 : 8;
    int k  = (elem < 8) ? (b0 + elem) : (16 + b0 + (elem - 8));
    int col = nt * 16 + (lane & 15);
    pw[t] = (_Float16)W[(kt * 32 + k) * F + col];
}

// ---------------- phase 1: h = x @ W via v_wmma_f32_16x16x32_f16 ----------------
// one wave computes one 16(M) x 128(N) strip of h; K=128 -> 4 wmma k-steps.
// B fragments staged once per block into LDS (32 KB of the 320 KB WGP pool).

__global__ void k_gemm(const _Float16* __restrict__ xh, const _Float16* __restrict__ pw,
                       float* __restrict__ h, int ntiles) {
    __shared__ __align__(32) _Float16 sW[PW_ELEMS];

    // cooperative copy: 2048 x 16B chunks over 128 threads (16 each)
    for (int i = threadIdx.x; i < PW_ELEMS / 8; i += blockDim.x)
        ((v8h*)sW)[i] = ((const v8h*)pw)[i];
    __syncthreads();

    int wave = threadIdx.x >> 5;
    int lane = threadIdx.x & 31;
    int tile = blockIdx.x * (blockDim.x >> 5) + wave;
    if (tile >= ntiles) return;

    int row = tile * 16 + (lane & 15);
    int b0  = (lane >> 4) << 3;          // 0 or 8
    const _Float16* xrow = xh + (size_t)row * F;

    v8f acc[8] = {};
#pragma unroll
    for (int kt = 0; kt < 4; ++kt) {
        const _Float16* pA = xrow + kt * 32;
        v8h alo = *(const v8h*)(pA + b0);          // K = b0 .. b0+7
        v8h ahi = *(const v8h*)(pA + 16 + b0);     // K = 16+b0 .. 16+b0+7
        v16h a = __builtin_shufflevector(alo, ahi,
                                         0, 1, 2, 3, 4, 5, 6, 7,
                                         8, 9, 10, 11, 12, 13, 14, 15);
#pragma unroll
        for (int nt = 0; nt < 8; ++nt) {
            v16h b = *(const v16h*)(sW + (((kt * 8 + nt) * 32 + lane) << 4));
            acc[nt] = __builtin_amdgcn_wmma_f32_16x16x32_f16(
                false, a, false, b, (short)0, acc[nt], false, false);
        }
    }

    int mbase = tile * 16 + ((lane >> 4) << 3);
    int col0  = lane & 15;
#pragma unroll
    for (int nt = 0; nt < 8; ++nt) {
#pragma unroll
        for (int r = 0; r < 8; ++r)
            h[(size_t)(mbase + r) * F + nt * 16 + col0] = acc[nt][r];
    }
}

// ---------------- phase 2: per-node logits, zero out accumulator ----------------

__global__ void k_node1(const float* __restrict__ h, const float* __restrict__ att_src,
                        const float* __restrict__ att_dst, float* __restrict__ asrc,
                        float* __restrict__ adst, float* __restrict__ out, int NH) {
    int t = blockIdx.x * blockDim.x + threadIdx.x;   // (node, head)
    if (t >= NH) return;
    int node = t >> 2, head = t & 3;
    const float* hp = h + (size_t)node * F + head * CH;
    const float* as = att_src + head * CH;
    const float* ad = att_dst + head * CH;
    float* op = out + (size_t)node * F + head * CH;
    float s = 0.0f, d = 0.0f;
#pragma unroll
    for (int c = 0; c < CH; ++c) {
        float v = hp[c];
        s += v * as[c];
        d += v * ad[c];
        op[c] = 0.0f;
    }
    asrc[t] = s;
    adst[t] = d;
}

// seed segment-max with the guaranteed self-loop logit (ordered-uint encoding)
__global__ void k_node2(const float* __restrict__ asrc, const float* __restrict__ adst,
                        unsigned* __restrict__ menc, int NH) {
    int t = blockIdx.x * blockDim.x + threadIdx.x;
    if (t >= NH) return;
    menc[t] = enc_ord(lrelu(asrc[t] + adst[t]));
}

// ---------------- phase 3: edge segment-softmax (3 passes) ----------------

__global__ void k_edge_max(const int* __restrict__ ei, const float* __restrict__ asrc,
                           const float* __restrict__ adst, unsigned* __restrict__ menc, int E) {
    int e = blockIdx.x * blockDim.x + threadIdx.x;
    if (e >= E) return;
    int s = ei[e], d = ei[E + e];
    float4 a = *(const float4*)(asrc + s * 4);
    float4 b = *(const float4*)(adst + d * 4);
    float ev[4] = {a.x + b.x, a.y + b.y, a.z + b.z, a.w + b.w};
#pragma unroll
    for (int q = 0; q < 4; ++q)
        atomicMax(menc + d * 4 + q, enc_ord(lrelu(ev[q])));
}

// decode max in place; seed denom with self-loop probability
__global__ void k_node3(const float* __restrict__ asrc, const float* __restrict__ adst,
                        unsigned* __restrict__ menc, float* __restrict__ denom, int NH) {
    int t = blockIdx.x * blockDim.x + threadIdx.x;
    if (t >= NH) return;
    float m = dec_ord(menc[t]);
    ((float*)menc)[t] = m;
    float e = lrelu(asrc[t] + adst[t]);
    denom[t] = __expf(e - m);
}

__global__ void k_edge_sum(const int* __restrict__ ei, const float* __restrict__ asrc,
                           const float* __restrict__ adst, const float* __restrict__ m,
                           float* __restrict__ denom, int E) {
    int e = blockIdx.x * blockDim.x + threadIdx.x;
    if (e >= E) return;
    int s = ei[e], d = ei[E + e];
    float4 a  = *(const float4*)(asrc + s * 4);
    float4 b  = *(const float4*)(adst + d * 4);
    float4 mm = *(const float4*)(m + d * 4);
    float ev[4] = {a.x + b.x, a.y + b.y, a.z + b.z, a.w + b.w};
    float mv[4] = {mm.x, mm.y, mm.z, mm.w};
#pragma unroll
    for (int q = 0; q < 4; ++q)
        atomicAdd(denom + d * 4 + q, __expf(lrelu(ev[q]) - mv[q]));
}

// one wave32 per edge; lane l owns channels {l, l+32, l+64, l+96} (head == q)
__global__ void k_edge_aggr(const int* __restrict__ ei, const float* __restrict__ asrc,
                            const float* __restrict__ adst, const float* __restrict__ m,
                            const float* __restrict__ denom, const float* __restrict__ h,
                            float* __restrict__ out, int E) {
    int lane = threadIdx.x & 31;
    int e = blockIdx.x * (blockDim.x >> 5) + (threadIdx.x >> 5);
    if (e >= E) return;
    int s = ei[e], d = ei[E + e];
    const float* hs = h + (size_t)s * F;
    // overlap the random 512B row gather with the alpha math: each lane
    // prefetches a disjoint 16B chunk -> whole row's cachelines in flight
    __builtin_prefetch(hs + lane * 4, 0, 3);
    float4 a  = *(const float4*)(asrc + s * 4);
    float4 b  = *(const float4*)(adst + d * 4);
    float4 mm = *(const float4*)(m + d * 4);
    float4 dn = *(const float4*)(denom + d * 4);
    float al[4];
    al[0] = __expf(lrelu(a.x + b.x) - mm.x) / (dn.x + GAT_EPS);
    al[1] = __expf(lrelu(a.y + b.y) - mm.y) / (dn.y + GAT_EPS);
    al[2] = __expf(lrelu(a.z + b.z) - mm.z) / (dn.z + GAT_EPS);
    al[3] = __expf(lrelu(a.w + b.w) - mm.w) / (dn.w + GAT_EPS);
    float* od = out + (size_t)d * F;
#pragma unroll
    for (int q = 0; q < 4; ++q)
        atomicAdd(od + q * 32 + lane, al[q] * hs[q * 32 + lane]);
}

// ---------------- phase 4: self-loop contribution, bias, exact GELU ----------------

__global__ void k_final(const float* __restrict__ h, const float* __restrict__ asrc,
                        const float* __restrict__ adst, const float* __restrict__ m,
                        const float* __restrict__ denom, const float* __restrict__ bias,
                        float* __restrict__ out, int total) {
    int t = blockIdx.x * blockDim.x + threadIdx.x;
    if (t >= total) return;
    int node = t >> 7;
    int ch   = t & 127;
    int head = ch >> 5;
    int nh   = node * 4 + head;
    float e = lrelu(asrc[nh] + adst[nh]);
    float alpha = __expf(e - m[nh]) / (denom[nh] + GAT_EPS);
    float v = out[t] + alpha * h[t] + bias[ch];
    out[t] = 0.5f * v * (1.0f + erff(v * 0.70710678118654752f));
}

// ---------------- host side ----------------

extern "C" void kernel_launch(void* const* d_in, const int* in_sizes, int n_in,
                              void* d_out, int out_size, void* d_ws, size_t ws_size,
                              hipStream_t stream) {
    const float* x       = (const float*)d_in[0];
    const int*   ei      = (const int*)d_in[1];
    const float* W       = (const float*)d_in[2];
    const float* att_src = (const float*)d_in[3];
    const float* att_dst = (const float*)d_in[4];
    const float* bias    = (const float*)d_in[5];
    float* out = (float*)d_out;

    const int N  = in_sizes[0] / F;     // 50000
    const int E  = in_sizes[1] / 2;     // 1,600,000
    const int NH = N * HEADS;
    const int Np = (N + 15) & ~15;      // pad rows to 16 for WMMA tiles

    char* ws = (char*)d_ws;
    size_t off = 0;
    auto take = [&](size_t bytes) -> void* {
        void* p = ws + off;
        off = (off + bytes + 255) & ~(size_t)255;
        return p;
    };
    _Float16* xh    = (_Float16*)take((size_t)Np * F * sizeof(_Float16));
    _Float16* pw    = (_Float16*)take((size_t)PW_ELEMS * sizeof(_Float16));
    float*    h     = (float*)take((size_t)Np * F * sizeof(float));
    float*    asrc  = (float*)take((size_t)NH * sizeof(float));
    float*    adst  = (float*)take((size_t)NH * sizeof(float));
    unsigned* menc  = (unsigned*)take((size_t)NH * sizeof(unsigned));  // later reused as float m
    float*    denom = (float*)take((size_t)NH * sizeof(float));
    (void)ws_size; (void)n_in; (void)out_size;

    const int T = 256;
    // phase 0
    k_cvt_x<<<(N * F + T - 1) / T, T, 0, stream>>>(x, xh, N * F);
    k_pack_w<<<(PW_ELEMS + T - 1) / T, T, 0, stream>>>(W, pw);
    // phase 1: WMMA GEMM (one wave per 16-row strip, 4 waves / 128-thread block)
    int ntiles = Np / 16;
    k_gemm<<<(ntiles + 3) / 4, 128, 0, stream>>>(xh, pw, h, ntiles);
    // phase 2
    k_node1<<<(NH + T - 1) / T, T, 0, stream>>>(h, att_src, att_dst, asrc, adst, out, NH);
    k_node2<<<(NH + T - 1) / T, T, 0, stream>>>(asrc, adst, menc, NH);
    // phase 3: segment softmax
    k_edge_max<<<(E + T - 1) / T, T, 0, stream>>>(ei, asrc, adst, menc, E);
    k_node3<<<(NH + T - 1) / T, T, 0, stream>>>(asrc, adst, menc, denom, NH);
    float* mf = (float*)menc;
    k_edge_sum<<<(E + T - 1) / T, T, 0, stream>>>(ei, asrc, adst, mf, denom, E);
    k_edge_aggr<<<(E + 7) / 8, T, 0, stream>>>(ei, asrc, adst, mf, denom, h, out, E);
    // phase 4
    k_final<<<(N * F + T - 1) / T, T, 0, stream>>>(h, asrc, adst, mf, denom, bias, out, N * F);
}
